// ExtBertEncoder_20057497272704
// MI455X (gfx1250) — matrix operand order repaired and checked by reading, not compiled, not run on hardware
//
#include <hip/hip_runtime.h>
#include <hip/hip_bf16.h>

// ---------------------------------------------------------------------------
// BERT encoder (L=12, B=8, S=512, H=768, NH=12, DH=64, I=3072) for gfx1250.
// bf16 WMMA everywhere; TDM (tensor_load_to_lds) double-buffers ALL shared
// tiles (GEMM A+B, attention K+V); V is transposed in LDS so every WMMA
// operand load is a vector ds_load.
// ---------------------------------------------------------------------------

#define L_  12
#define B_  8
#define S_  512
#define H_  768
#define NH_ 12
#define DH_ 64
#define I_  3072
#define MTOK (B_ * S_)       // 4096 token rows
#define H3  (3 * H_)         // 2304

typedef __bf16 bf16_t;
typedef __attribute__((ext_vector_type(16))) __bf16 v16bf;
typedef __attribute__((ext_vector_type(8)))  __bf16 v8bf;
typedef __attribute__((ext_vector_type(2)))  __bf16 v2bf;
typedef __attribute__((ext_vector_type(8)))  float  v8f;
typedef __attribute__((ext_vector_type(4)))  unsigned int v4u;
typedef __attribute__((ext_vector_type(8)))  int v8i;
typedef __attribute__((ext_vector_type(4)))  int v4i;

// ---- WMMA wrapper ----------------------------------------------------------
__device__ __forceinline__ v8f wmma_bf16(v16bf a, v16bf b, v8f c) {
  return __builtin_amdgcn_wmma_f32_16x16x32_bf16(false, a, false, b,
                                                 (short)0, c, false, false);
}

// ---- Fragment loaders (ISA §7.12.2 layouts, wave32), bf16 in memory/LDS ----
// A (16x32): lane = row M=lane%16; elements 0..7 at K=k0+8*half,
//            elements 8..15 at K=k0+16+8*half (two 16B chunks).
__device__ __forceinline__ v16bf load_af16(const bf16_t* rowp, int k0, int half) {
  v8bf lo = *reinterpret_cast<const v8bf*>(rowp + k0 + 8 * half);
  v8bf hi = *reinterpret_cast<const v8bf*>(rowp + k0 + 16 + 8 * half);
  return __builtin_shufflevector(lo, hi, 0, 1, 2, 3, 4, 5, 6, 7,
                                 8, 9, 10, 11, 12, 13, 14, 15);
}
// B (32x16): lane = column N=lane%16 (a row of W[N,K]); elements 0..7 at
//            K=k0+8*half, elements 8..15 at K=k0+16+8*half (two 16B chunks).
__device__ __forceinline__ v16bf load_bf16v(const bf16_t* rowp, int k0, int half) {
  v8bf lo = *reinterpret_cast<const v8bf*>(rowp + k0 + 8 * half);
  v8bf hi = *reinterpret_cast<const v8bf*>(rowp + k0 + 16 + 8 * half);
  return __builtin_shufflevector(lo, hi, 0, 1, 2, 3, 4, 5, 6, 7,
                                 8, 9, 10, 11, 12, 13, 14, 15);
}

__device__ __forceinline__ float gelu_exact(float v) {
  return 0.5f * v * (1.0f + erff(v * 0.70710678118654752f));
}

__device__ __forceinline__ unsigned lds_off(const void* p) {
  return (unsigned)(size_t)p;  // generic LDS addr: low 32 bits = LDS byte offset
}

// ---- TDM: 2D tile (width elems x rows, 2B elements) from global
//      (row stride `stride` elems) into packed LDS rows. D# per ISA §8.3/§8.4.
__device__ __forceinline__ void tdm_issue_2d(const bf16_t* gptr, unsigned width,
                                             unsigned rows, unsigned stride,
                                             unsigned ldsoff) {
  const unsigned long long ga = (unsigned long long)(size_t)gptr;
  v4u g0;
  g0[0] = 1u;                                    // count=1 (valid), user mode
  g0[1] = ldsoff;                                // lds_addr (bytes)
  g0[2] = (unsigned)ga;                          // global_addr[31:0]
  g0[3] = (unsigned)((ga >> 32) & 0x1FFFFFFull)  // global_addr[56:32]
          | (2u << 30);                          // type=2 ("image")
  v8i g1;
  g1[0] = 0x00010000;                            // data_size=1 -> 2-byte elems
  g1[1] = (int)((width & 0xFFFFu) << 16);        // tensor_dim0[15:0]
  g1[2] = (int)(((width >> 16) & 0xFFFFu)        // tensor_dim0[31:16]
          | (rows << 16));                       // tensor_dim1[15:0]
  g1[3] = (int)(width << 16);                    // tensor_dim1 hi=0 | tile_dim0
  g1[4] = (int)rows;                             // tile_dim1 | tile_dim2=0
  g1[5] = (int)stride;                           // tensor_dim0_stride[31:0]
  g1[6] = 0;                                     // stride hi | dim1_stride lo
  g1[7] = 0;
  v4i z4 = {0, 0, 0, 0};
#if defined(__clang_major__) && (__clang_major__ >= 23)
  v8i z8 = {0, 0, 0, 0, 0, 0, 0, 0};
  __builtin_amdgcn_tensor_load_to_lds(g0, g1, z4, z4, z8, 0);
#else
  __builtin_amdgcn_tensor_load_to_lds(g0, g1, z4, z4, 0);
#endif
}

// ---------------------------------------------------------------------------
// GEMM: C[M,N] = A[M,K] @ W[N,K]^T + bias  (A,W bf16; acc fp32)
// 256 threads = 8 waves (4 M x 2 N); block tile 128M x 128N; wave 32M x 64N.
// Both A (128x32) and B (128x32) tiles TDM-staged in LDS, double buffered:
// inner loop is ds_load + 8x v_wmma per 32-K step.
// ---------------------------------------------------------------------------
template <int GELU, int OUT16>
__global__ __launch_bounds__(256) void gemm_wmma(
    const bf16_t* __restrict__ A, const bf16_t* __restrict__ W,
    const float* __restrict__ bias, float* __restrict__ Cf,
    bf16_t* __restrict__ Cb, int M, int N, int K) {
  __shared__ alignas(32) bf16_t ldsA[2][128 * 32];
  __shared__ alignas(32) bf16_t ldsW[2][128 * 32];

  const int wave = threadIdx.x >> 5, lane = threadIdx.x & 31;
  const int lm = lane & 15, half = lane >> 4;
  const int wm = wave & 3, wn = wave >> 2;
  const int mblk = blockIdx.y * 128;
  const int nblk = blockIdx.x * 128;
  const int m0 = mblk + wm * 32;
  const int n0 = nblk + wn * 64;
  const int nk = K >> 5;

  const bf16_t* abase = A + (size_t)mblk * K;
  const bf16_t* wbase = W + (size_t)nblk * K;

  if (wave == 0) {
    tdm_issue_2d(abase, 32, 128, (unsigned)K, lds_off(&ldsA[0][0]));
    tdm_issue_2d(wbase, 32, 128, (unsigned)K, lds_off(&ldsW[0][0]));
  }

  v8f acc[2][4] = {};
  for (int i = 0; i < nk; ++i) {
    if (wave == 0) __builtin_amdgcn_s_wait_tensorcnt((short)0);
    __syncthreads();  // buf i&1 ready; all waves done reading buf (i-1)&1
    if (wave == 0 && i + 1 < nk) {
      const int k1 = (i + 1) << 5;
      tdm_issue_2d(abase + k1, 32, 128, (unsigned)K,
                   lds_off(&ldsA[(i + 1) & 1][0]));
      tdm_issue_2d(wbase + k1, 32, 128, (unsigned)K,
                   lds_off(&ldsW[(i + 1) & 1][0]));
    }
    const bf16_t* la = &ldsA[i & 1][0];
    const bf16_t* lb = &ldsW[i & 1][0];
    v16bf af0 = load_af16(la + (size_t)(wm * 32 + lm) * 32, 0, half);
    v16bf af1 = load_af16(la + (size_t)(wm * 32 + 16 + lm) * 32, 0, half);
#pragma unroll
    for (int t = 0; t < 4; ++t) {
      v16bf bf = load_bf16v(lb + (size_t)(wn * 64 + t * 16 + lm) * 32, 0, half);
      acc[0][t] = wmma_bf16(af0, bf, acc[0][t]);
      acc[1][t] = wmma_bf16(af1, bf, acc[1][t]);
    }
  }

#pragma unroll
  for (int u = 0; u < 2; ++u)
#pragma unroll
    for (int t = 0; t < 4; ++t) {
      const int n = n0 + t * 16 + lm;
      const float bv = bias[n];
#pragma unroll
      for (int r = 0; r < 8; ++r) {
        float v = acc[u][t][r] + bv;
        if (GELU) v = gelu_exact(v);
        const size_t idx = (size_t)(m0 + u * 16 + r + 8 * half) * N + n;
        if (OUT16) Cb[idx] = (bf16_t)v;
        else       Cf[idx] = v;
      }
    }
}

// ---------------------------------------------------------------------------
// Attention: single-pass flash per (b, h); 4 waves = 4 q-tiles share TDM-staged
// K/V tiles (32 keys x 64 dh, double buffered). V is cooperatively transposed
// in LDS so P.V B-fragments are contiguous vector ds_loads.
// ---------------------------------------------------------------------------
__global__ __launch_bounds__(128) void attn_kernel(
    const bf16_t* __restrict__ qkv, const float* __restrict__ mask,
    bf16_t* __restrict__ ctx) {
  __shared__ alignas(32) bf16_t ldsK[2][32 * 64];
  __shared__ alignas(32) bf16_t ldsV[2][32 * 64];   // key-major (TDM dest)
  __shared__ alignas(32) bf16_t ldsVt[64 * 32];     // dh-major (transposed)
  __shared__ alignas(32) bf16_t pls[4][16][32];     // per-wave p-tile transpose

  const int tid = threadIdx.x;
  const int wave = tid >> 5;
  const int lane = tid & 31;
  const int lm = lane & 15, half = lane >> 4;
  const int b = blockIdx.z, h = blockIdx.y;
  const int qt = blockIdx.x * 4 + wave;
  const float scale = 0.125f;  // 1/sqrt(64)

  const bf16_t* qbase = qkv + (size_t)(b * S_) * H3 + h * DH_;
  const bf16_t* kbase = qbase + H_;
  const bf16_t* vbase = qbase + 2 * H_;
  const float* mrow = mask + (size_t)b * S_;

  const int sq = qt * 16 + lm;
  const bf16_t* qrow = qbase + (size_t)sq * H3;
  v16bf qf[2];
#pragma unroll
  for (int kk = 0; kk < 2; ++kk) qf[kk] = load_af16(qrow, kk * 32, half);

  if (wave == 0) {
    tdm_issue_2d(kbase, 64, 32, H3, lds_off(&ldsK[0][0]));
    tdm_issue_2d(vbase, 64, 32, H3, lds_off(&ldsV[0][0]));
  }

  // V-transpose task split: 128 threads = 16 key-pairs x 8 dh-chunks.
  const int kp = tid & 15;   // keys 2*kp, 2*kp+1
  const int dc = tid >> 4;   // dh chunk dc*8 .. dc*8+7

  float m[8], l[8];
#pragma unroll
  for (int r = 0; r < 8; ++r) { m[r] = -1e30f; l[r] = 0.f; }

  v8f cacc[4] = {};
  for (int i = 0; i < S_ / 32; ++i) {
    const int k0 = i * 32;
    if (wave == 0) __builtin_amdgcn_s_wait_tensorcnt((short)0);
    __syncthreads();  // K/V buf i&1 ready; prev P.V reads of ldsVt done
    if (wave == 0 && i + 1 < S_ / 32) {
      const bf16_t* knext = kbase + (size_t)(k0 + 32) * H3;
      const bf16_t* vnext = vbase + (size_t)(k0 + 32) * H3;
      tdm_issue_2d(knext, 64, 32, H3, lds_off(&ldsK[(i + 1) & 1][0]));
      tdm_issue_2d(vnext, 64, 32, H3, lds_off(&ldsV[(i + 1) & 1][0]));
    }
    const bf16_t* lk = &ldsK[i & 1][0];
    const bf16_t* lv = &ldsV[i & 1][0];

    // cooperative V transpose: key-major -> dh-major, packed 2-key stores
    {
      v8bf r0 = *reinterpret_cast<const v8bf*>(lv + (size_t)(2 * kp) * 64 + dc * 8);
      v8bf r1 = *reinterpret_cast<const v8bf*>(lv + (size_t)(2 * kp + 1) * 64 + dc * 8);
#pragma unroll
      for (int j = 0; j < 8; ++j) {
        v2bf t2;
        t2[0] = r0[j];
        t2[1] = r1[j];
        *reinterpret_cast<v2bf*>(&ldsVt[(size_t)(dc * 8 + j) * 32 + 2 * kp]) = t2;
      }
    }

    // two 16-key score tiles (C layout: elem r -> row r+8*half, lane -> key)
    v8f a0 = {}, a1 = {};
#pragma unroll
    for (int kk = 0; kk < 2; ++kk) {
      a0 = wmma_bf16(qf[kk], load_bf16v(lk + (size_t)lm * 64, kk * 32, half), a0);
      a1 = wmma_bf16(qf[kk], load_bf16v(lk + (size_t)(16 + lm) * 64, kk * 32, half), a1);
    }
    const float mk0 = mrow[k0 + lm];
    const float mk1 = mrow[k0 + 16 + lm];

    float corrf[8];
#pragma unroll
    for (int r = 0; r < 8; ++r) {
      const float s0 = a0[r] * scale + mk0;
      const float s1 = a1[r] * scale + mk1;
      float tm = fmaxf(s0, s1);
#pragma unroll
      for (int off = 1; off < 16; off <<= 1) tm = fmaxf(tm, __shfl_xor(tm, off));
      const float mnew = fmaxf(m[r], tm);
      corrf[r] = __expf(m[r] - mnew);
      m[r] = mnew;
      const float p0 = __expf(s0 - mnew);
      const float p1 = __expf(s1 - mnew);
      float es = p0 + p1;
#pragma unroll
      for (int off = 1; off < 16; off <<= 1) es += __shfl_xor(es, off);
      l[r] = l[r] * corrf[r] + es;
      pls[wave][r + 8 * half][lm]      = (bf16_t)p0;  // key col k0+lm
      pls[wave][r + 8 * half][16 + lm] = (bf16_t)p1;  // key col k0+16+lm
    }
    // rescale running context
#pragma unroll
    for (int t = 0; t < 4; ++t)
#pragma unroll
      for (int r = 0; r < 8; ++r) cacc[t][r] *= corrf[r];

    __syncthreads();  // ldsVt fully written (cross-wave)

    // p-tile round-trip is wave-private; same-wave LDS ops are in-order.
    v16bf pf = load_af16(&pls[wave][lm][0], 0, half);
#pragma unroll
    for (int t = 0; t < 4; ++t) {  // P.V: contiguous B-fragments from ldsVt
      v16bf bf = load_bf16v(ldsVt + (size_t)(t * 16 + lm) * 32, 0, half);
      cacc[t] = wmma_bf16(pf, bf, cacc[t]);
    }
  }

#pragma unroll
  for (int t = 0; t < 4; ++t) {
    const int d = t * 16 + lm;
#pragma unroll
    for (int r = 0; r < 8; ++r) {
      const int row = qt * 16 + r + 8 * half;
      ctx[(size_t)(b * S_ + row) * H_ + h * DH_ + d] = (bf16_t)(cacc[t][r] / l[r]);
    }
  }
}

// ---------------------------------------------------------------------------
// Residual add + LayerNorm; writes fp32 x (residual stream) + bf16 copy.
// ---------------------------------------------------------------------------
__global__ __launch_bounds__(256) void add_ln_kernel(
    float* __restrict__ x, const float* __restrict__ y,
    const float* __restrict__ g, const float* __restrict__ b,
    bf16_t* __restrict__ xb) {
  __shared__ float red[256];
  const int row = blockIdx.x, t = threadIdx.x;
  float* xr = x + (size_t)row * H_;
  const float* yr = y + (size_t)row * H_;
  bf16_t* xbr = xb + (size_t)row * H_;

  float v[3];
  float s = 0.f;
#pragma unroll
  for (int j = 0; j < 3; ++j) {
    const int c = t + 256 * j;
    v[j] = xr[c] + yr[c];
    s += v[j];
  }
  red[t] = s;
  __syncthreads();
  for (int o = 128; o > 0; o >>= 1) {
    if (t < o) red[t] += red[t + o];
    __syncthreads();
  }
  const float mean = red[0] * (1.f / (float)H_);
  __syncthreads();

  float vs = 0.f;
#pragma unroll
  for (int j = 0; j < 3; ++j) {
    const float d = v[j] - mean;
    vs += d * d;
  }
  red[t] = vs;
  __syncthreads();
  for (int o = 128; o > 0; o >>= 1) {
    if (t < o) red[t] += red[t + o];
    __syncthreads();
  }
  const float rstd = rsqrtf(red[0] * (1.f / (float)H_) + 1e-12f);
#pragma unroll
  for (int j = 0; j < 3; ++j) {
    const int c = t + 256 * j;
    const float o = (v[j] - mean) * rstd * g[c] + b[c];
    xr[c] = o;
    xbr[c] = (bf16_t)o;
  }
}

// ---------------------------------------------------------------------------
// fp32 -> bf16 conversion (8 elems/thread) and x/xb16 init.
// ---------------------------------------------------------------------------
__global__ __launch_bounds__(256) void cvt_bf16_kernel(
    const float* __restrict__ s, bf16_t* __restrict__ d) {
  const size_t i = ((size_t)blockIdx.x * 256 + threadIdx.x) * 8;
  const float4 a = *reinterpret_cast<const float4*>(s + i);
  const float4 c = *reinterpret_cast<const float4*>(s + i + 4);
  v8bf o;
  o[0] = (bf16_t)a.x; o[1] = (bf16_t)a.y; o[2] = (bf16_t)a.z; o[3] = (bf16_t)a.w;
  o[4] = (bf16_t)c.x; o[5] = (bf16_t)c.y; o[6] = (bf16_t)c.z; o[7] = (bf16_t)c.w;
  *reinterpret_cast<v8bf*>(d + i) = o;
}

__global__ __launch_bounds__(256) void init_x_kernel(
    const float* __restrict__ h, float* __restrict__ x, bf16_t* __restrict__ xb) {
  const size_t i = (size_t)blockIdx.x * 256 + threadIdx.x;
  const float v = h[i];
  x[i] = v;
  xb[i] = (bf16_t)v;
}

// ---------------------------------------------------------------------------
// Host orchestration
// ---------------------------------------------------------------------------
extern "C" void kernel_launch(void* const* d_in, const int* in_sizes, int n_in,
                              void* d_out, int out_size, void* d_ws, size_t ws_size,
                              hipStream_t stream) {
  (void)in_sizes; (void)n_in; (void)out_size; (void)ws_size;

  const float* hidden = (const float*)d_in[0];
  const float* mask   = (const float*)d_in[1];
  const float* gA = (const float*)d_in[2];
  const float* bA = (const float*)d_in[3];
  const float* gB = (const float*)d_in[4];
  const float* bB = (const float*)d_in[5];
  const float* wa = (const float*)d_in[6];
  const float* ba = (const float*)d_in[7];
  const float* wb = (const float*)d_in[8];
  const float* bb = (const float*)d_in[9];
  const float* wc = (const float*)d_in[10];
  const float* bc = (const float*)d_in[11];
  const float* wd = (const float*)d_in[12];
  const float* bd = (const float*)d_in[13];

  float*  x     = (float*)d_out;                     // residual stream [MTOK,H] f32
  float*  tmp   = (float*)d_ws;                      // [MTOK,H] f32
  bf16_t* xb16  = (bf16_t*)(tmp + (size_t)MTOK * H_);
  bf16_t* qkv16 = xb16  + (size_t)MTOK * H_;
  bf16_t* ctx16 = qkv16 + (size_t)MTOK * H3;
  bf16_t* h16   = ctx16 + (size_t)MTOK * H_;
  bf16_t* wa16  = h16   + (size_t)MTOK * I_;
  bf16_t* wb16  = wa16  + (size_t)H3 * H_;
  bf16_t* wc16  = wb16  + (size_t)H_ * H_;
  bf16_t* wd16  = wc16  + (size_t)I_ * H_;

  init_x_kernel<<<dim3(MTOK * H_ / 256), dim3(256), 0, stream>>>(hidden, x, xb16);

  for (int l = 0; l < L_; ++l) {
    const float* ba_l = ba + (size_t)l * H3;
    const float* bb_l = bb + (size_t)l * H_;
    const float* bc_l = bc + (size_t)l * I_;
    const float* bd_l = bd + (size_t)l * H_;

    // per-layer weight bf16 images
    cvt_bf16_kernel<<<dim3(H3 * H_ / 2048), dim3(256), 0, stream>>>(
        wa + (size_t)l * H3 * H_, wa16);
    cvt_bf16_kernel<<<dim3(H_ * H_ / 2048), dim3(256), 0, stream>>>(
        wb + (size_t)l * H_ * H_, wb16);
    cvt_bf16_kernel<<<dim3(I_ * H_ / 2048), dim3(256), 0, stream>>>(
        wc + (size_t)l * I_ * H_, wc16);
    cvt_bf16_kernel<<<dim3(H_ * I_ / 2048), dim3(256), 0, stream>>>(
        wd + (size_t)l * H_ * I_, wd16);

    // fused QKV projection (bf16 out)
    gemm_wmma<0, 1><<<dim3(H3 / 128, MTOK / 128), dim3(256), 0, stream>>>(
        xb16, wa16, ba_l, nullptr, qkv16, MTOK, H3, H_);
    // attention core (bf16 ctx out)
    attn_kernel<<<dim3(S_ / 64, NH_, B_), dim3(128), 0, stream>>>(
        qkv16, mask, ctx16);
    // output projection (f32 out for residual)
    gemm_wmma<0, 0><<<dim3(H_ / 128, MTOK / 128), dim3(256), 0, stream>>>(
        ctx16, wb16, bb_l, tmp, nullptr, MTOK, H_, H_);
    // residual + post-LN (A), refresh bf16 activations
    add_ln_kernel<<<dim3(MTOK), dim3(256), 0, stream>>>(
        x, tmp, gA + (size_t)l * H_, bA + (size_t)l * H_, xb16);
    // FFN up + exact GELU (bf16 out)
    gemm_wmma<1, 1><<<dim3(I_ / 128, MTOK / 128), dim3(256), 0, stream>>>(
        xb16, wc16, bc_l, nullptr, h16, MTOK, I_, H_);
    // FFN down (f32 out)
    gemm_wmma<0, 0><<<dim3(H_ / 128, MTOK / 128), dim3(256), 0, stream>>>(
        h16, wd16, bd_l, tmp, nullptr, MTOK, H_, I_);
    // residual + post-LN (B)
    add_ln_kernel<<<dim3(MTOK), dim3(256), 0, stream>>>(
        x, tmp, gB + (size_t)l * H_, bB + (size_t)l * H_, xb16);
  }
}
